// Linear_25211458027737
// MI455X (gfx1250) — matrix operand order repaired and checked by reading, not compiled, not run on hardware
//
#include <hip/hip_runtime.h>
#include <hip/hip_bf16.h>

// Grouped-LoRA fused linear for MI455X (gfx1250), wave32 + bf16 WMMA.
// out[m,n] = sum_k x[m,k]W[n,k] + b[n] + 2.0 * sum_r (sum_k x[m,k]A_g[r,k]) * B_g[n,r]
// f32 inputs split into bf16 hi/lo; 3 bf16 WMMAs approximate the f32 product.
// k-loop is software-pipelined: next tile's global loads issue before this
// tile's WMMA block, so HBM/L2 latency hides behind matrix work.

#define BM 128
#define BN 128
#define BK 32
#define D_DIM 4096
#define O_DIM 4096
#define M_DIM 16384
#define R_DIM 16
#define SEG_ROWS 4096   // rows per adapter segment: (B/G)*S = 2*2048
#define LDSK 48         // padded LDS row stride (elements): 96B rows, 16B-aligned frags

typedef __attribute__((ext_vector_type(16))) __bf16 v16bf;
typedef __attribute__((ext_vector_type(16))) unsigned short u16x16;
typedef __attribute__((ext_vector_type(8))) float v8f;

__device__ __forceinline__ unsigned short f32_to_bf16_rn(float f) {
  unsigned int u = __float_as_uint(f);
  u += 0x7fffu + ((u >> 16) & 1u);
  return (unsigned short)(u >> 16);
}
__device__ __forceinline__ void split_store(float f, unsigned short* hi, unsigned short* lo) {
  unsigned short h = f32_to_bf16_rn(f);
  *hi = h;
  // lo: exact f32 residual, truncated to bf16 (residual rounding ~1e-5 rel; negligible)
  float r = f - __uint_as_float(((unsigned int)h) << 16);
  *lo = (unsigned short)(__float_as_uint(r) >> 16);
}

__global__ __launch_bounds__(256)
void lora_linear_wmma(const float* __restrict__ x,
                      const float* __restrict__ W,
                      const float* __restrict__ bias,
                      const float* __restrict__ lora_a,
                      const float* __restrict__ lora_b,
                      float* __restrict__ out) {
  __shared__ unsigned short sXh[BM][LDSK];
  __shared__ unsigned short sXl[BM][LDSK];
  __shared__ unsigned short sWh[BN][LDSK];
  __shared__ unsigned short sWl[BN][LDSK];
  __shared__ unsigned short sAh[R_DIM][LDSK];
  __shared__ unsigned short sAl[R_DIM][LDSK];
  __shared__ float sH[BM][R_DIM];

  const int tid    = threadIdx.x;
  const int lane   = tid & 31;
  const int wid    = tid >> 5;
  const int wave_m = wid & 3;    // 0..3 : 32-row slice of the 128-row tile
  const int wave_n = wid >> 2;   // 0..1 : 64-col slice of the 128-col tile
  const int half   = lane >> 4;  // lane group (VGPR-layout half)
  const int l16    = lane & 15;

  const int m0 = blockIdx.y * BM;
  const int n0 = blockIdx.x * BN;
  const int g  = m0 / SEG_ROWS; // adapter index; uniform per block (128 | 4096)

  const float* Ag = lora_a + (size_t)g * R_DIM * D_DIM;
  const float* Bg = lora_b + (size_t)g * O_DIM * R_DIM;

  // ---- per-thread staging coordinates (fixed across k-steps) ----
  int srow[4], scol[4];
#pragma unroll
  for (int it = 0; it < 4; ++it) {
    int idx = tid + it * 256;     // 0..1023
    srow[it] = idx >> 3;          // 0..127
    scol[it] = (idx & 7) << 2;    // 0,4,...,28
  }
  const bool doA   = (tid < 128);
  const int  arow  = tid >> 3;          // 0..15 (valid when doA)
  const int  acol  = (tid & 7) << 2;    // 0..28

  const float* xp[4];
  const float* wp[4];
#pragma unroll
  for (int it = 0; it < 4; ++it) {
    xp[it] = x + (size_t)(m0 + srow[it]) * D_DIM + scol[it];
    wp[it] = W + (size_t)(n0 + srow[it]) * D_DIM + scol[it];
  }
  const float* ap = Ag + (size_t)arow * D_DIM + acol;

  v8f acc[2][4];
  v8f hacc[2];
#pragma unroll
  for (int t = 0; t < 2; ++t) {
    hacc[t] = {};
#pragma unroll
    for (int j = 0; j < 4; ++j) acc[t][j] = {};
  }

  // ---- pipeline prologue: fetch k-tile 0 into registers ----
  float4 px[4], pw[4], pa;
#pragma unroll
  for (int it = 0; it < 4; ++it) {
    px[it] = *(const float4*)(xp[it]);
    pw[it] = *(const float4*)(wp[it]);
  }
  if (doA) pa = *(const float4*)(ap);

  for (int kb = 0; kb < D_DIM; kb += BK) {
    __syncthreads();
    // ---- drain staged registers -> bf16 hi/lo in LDS ----
#pragma unroll
    for (int it = 0; it < 4; ++it) {
      int row = srow[it], c4 = scol[it];
      split_store(px[it].x, &sXh[row][c4 + 0], &sXl[row][c4 + 0]);
      split_store(px[it].y, &sXh[row][c4 + 1], &sXl[row][c4 + 1]);
      split_store(px[it].z, &sXh[row][c4 + 2], &sXl[row][c4 + 2]);
      split_store(px[it].w, &sXh[row][c4 + 3], &sXl[row][c4 + 3]);
      split_store(pw[it].x, &sWh[row][c4 + 0], &sWl[row][c4 + 0]);
      split_store(pw[it].y, &sWh[row][c4 + 1], &sWl[row][c4 + 1]);
      split_store(pw[it].z, &sWh[row][c4 + 2], &sWl[row][c4 + 2]);
      split_store(pw[it].w, &sWh[row][c4 + 3], &sWl[row][c4 + 3]);
    }
    if (doA) {
      split_store(pa.x, &sAh[arow][acol + 0], &sAl[arow][acol + 0]);
      split_store(pa.y, &sAh[arow][acol + 1], &sAl[arow][acol + 1]);
      split_store(pa.z, &sAh[arow][acol + 2], &sAl[arow][acol + 2]);
      split_store(pa.w, &sAh[arow][acol + 3], &sAl[arow][acol + 3]);
    }
    __syncthreads();

    // ---- issue next tile's global loads NOW; waits land next iteration ----
    const int kn = kb + BK;
    if (kn < D_DIM) {
#pragma unroll
      for (int it = 0; it < 4; ++it) {
        px[it] = *(const float4*)(xp[it] + kn);
        pw[it] = *(const float4*)(wp[it] + kn);
      }
      if (doA) pa = *(const float4*)(ap + kn);
    }

    // ---- A fragments (16x32 bf16, contiguous layout: k = half*16 + i per lane) ----
    u16x16 axh[2], axl[2];
#pragma unroll
    for (int t = 0; t < 2; ++t) {
      int row = wave_m * 32 + t * 16 + l16;
#pragma unroll
      for (int i = 0; i < 8; ++i) {
        axh[t][i]     = sXh[row][half * 16 + i];
        axh[t][8 + i] = sXh[row][half * 16 + 8 + i];
        axl[t][i]     = sXl[row][half * 16 + i];
        axl[t][8 + i] = sXl[row][half * 16 + 8 + i];
      }
    }

    // ---- main GEMM: 2x4 WMMA tiles, 3 bf16 products per tile ----
#pragma unroll
    for (int j = 0; j < 4; ++j) {
      int row = wave_n * 64 + j * 16 + l16;   // W row = output feature n
      u16x16 bh, bl;
#pragma unroll
      for (int i = 0; i < 16; ++i) {
        bh[i] = sWh[row][half * 16 + i];
        bl[i] = sWl[row][half * 16 + i];
      }
      v16bf vbh = __builtin_bit_cast(v16bf, bh);
      v16bf vbl = __builtin_bit_cast(v16bf, bl);
#pragma unroll
      for (int t = 0; t < 2; ++t) {
        v16bf vah = __builtin_bit_cast(v16bf, axh[t]);
        v16bf val = __builtin_bit_cast(v16bf, axl[t]);
        acc[t][j] = __builtin_amdgcn_wmma_f32_16x16x32_bf16(
            false, vah, false, vbh, (short)0, acc[t][j], false, false);
        acc[t][j] = __builtin_amdgcn_wmma_f32_16x16x32_bf16(
            false, vah, false, vbl, (short)0, acc[t][j], false, false);
        acc[t][j] = __builtin_amdgcn_wmma_f32_16x16x32_bf16(
            false, val, false, vbh, (short)0, acc[t][j], false, false);
      }
    }

    // ---- LoRA h = x @ A_g^T : only wave_n==0 (wave-uniform; EXEC all-ones inside) ----
    if (wave_n == 0) {
      u16x16 agh, agl;
#pragma unroll
      for (int i = 0; i < 16; ++i) {
        agh[i] = sAh[l16][half * 16 + i];
        agl[i] = sAl[l16][half * 16 + i];
      }
      v16bf vgh = __builtin_bit_cast(v16bf, agh);
      v16bf vgl = __builtin_bit_cast(v16bf, agl);
#pragma unroll
      for (int t = 0; t < 2; ++t) {
        v16bf vah = __builtin_bit_cast(v16bf, axh[t]);
        v16bf val = __builtin_bit_cast(v16bf, axl[t]);
        hacc[t] = __builtin_amdgcn_wmma_f32_16x16x32_bf16(
            false, vah, false, vgh, (short)0, hacc[t], false, false);
        hacc[t] = __builtin_amdgcn_wmma_f32_16x16x32_bf16(
            false, vah, false, vgl, (short)0, hacc[t], false, false);
        hacc[t] = __builtin_amdgcn_wmma_f32_16x16x32_bf16(
            false, val, false, vgh, (short)0, hacc[t], false, false);
      }
    }
  }

  // ---- publish h tile (C-layout: VGPR v, lane -> M = v + half*8, r = l16) ----
  __syncthreads();
  if (wave_n == 0) {
#pragma unroll
    for (int t = 0; t < 2; ++t) {
      int mloc = wave_m * 32 + t * 16 + half * 8;
#pragma unroll
      for (int v = 0; v < 8; ++v) sH[mloc + v][l16] = hacc[t][v];
    }
  }
  __syncthreads();

  // ---- epilogue: bias + 2.0 * h . B_g[n,:]  (16 FMAs/element, runs once) ----
#pragma unroll
  for (int j = 0; j < 4; ++j) {
    int n = n0 + wave_n * 64 + j * 16 + l16;
    float bn = bias[n];
    const float* bgp = Bg + (size_t)n * R_DIM;
    float bg[16];
#pragma unroll
    for (int r = 0; r < 16; ++r) bg[r] = bgp[r];
#pragma unroll
    for (int t = 0; t < 2; ++t) {
#pragma unroll
      for (int v = 0; v < 8; ++v) {
        int mloc = wave_m * 32 + t * 16 + half * 8 + v;
        float dot = 0.f;
#pragma unroll
        for (int r = 0; r < 16; ++r) dot += sH[mloc][r] * bg[r];
        out[(size_t)(m0 + mloc) * O_DIM + n] = acc[t][j][v] + bn + 2.0f * dot;
      }
    }
  }
}

extern "C" void kernel_launch(void* const* d_in, const int* in_sizes, int n_in,
                              void* d_out, int out_size, void* d_ws, size_t ws_size,
                              hipStream_t stream) {
  (void)in_sizes; (void)n_in; (void)d_ws; (void)ws_size; (void)out_size;
  const float* x      = (const float*)d_in[0];
  const float* W      = (const float*)d_in[1];
  const float* b      = (const float*)d_in[2];
  const float* lora_a = (const float*)d_in[3];
  const float* lora_b = (const float*)d_in[4];
  float* out = (float*)d_out;

  dim3 grid(O_DIM / BN, M_DIM / BM);   // 32 x 128 = 4096 blocks
  dim3 block(256);                     // 8 wave32 waves
  lora_linear_wmma<<<grid, block, 0, stream>>>(x, W, b, lora_a, lora_b, out);
}
